// CombinedLoss_74758200754291
// MI455X (gfx1250) — matrix operand order repaired and checked by reading, not compiled, not run on hardware
//
#include <hip/hip_runtime.h>
#include <hip/hip_bf16.h>
#include <math.h>

// Problem constants (from reference)
#define Bc 16
#define Cc 512
#define Tc 1024
#define Vc 4096
#define Kc 100
#define Nc (Bc * Tc)   // 16384 tokens

typedef __attribute__((ext_vector_type(16))) __bf16 v16bf;
typedef __attribute__((ext_vector_type(8)))  __bf16 v8bf;
typedef __attribute__((ext_vector_type(8)))  float  v8f;

// ---------------------------------------------------------------------------
// Prep: elementwise f32 -> bf16 for W and codebook (same shape V x C)
// ---------------------------------------------------------------------------
__global__ void prep_convert(const float* __restrict__ W,
                             const float* __restrict__ CB,
                             __bf16* __restrict__ Wbf,
                             __bf16* __restrict__ CBbf) {
  int i = blockIdx.x * blockDim.x + threadIdx.x;
  if (i < Vc * Cc) {
    Wbf[i]  = (__bf16)W[i];
    CBbf[i] = (__bf16)CB[i];
  }
}

// ---------------------------------------------------------------------------
// Prep: (B, C, T) f32 -> (N=B*T, C) bf16 via coalesced 32x32 LDS tile transpose
// ---------------------------------------------------------------------------
__global__ void prep_transpose(const float* __restrict__ emb,
                               __bf16* __restrict__ embBf) {
  __shared__ float tile[32][33];
  int b  = blockIdx.z;
  int c0 = blockIdx.y * 32;
  int t0 = blockIdx.x * 32;
  int tid = threadIdx.x;
  int tl = tid & 31, g = tid >> 5;          // 8 row-groups of the tile
  const float* src = emb + (size_t)b * Cc * Tc;
  #pragma unroll
  for (int i = 0; i < 4; ++i) {
    int cl = g + i * 8;
    tile[cl][tl] = src[(size_t)(c0 + cl) * Tc + (t0 + tl)];   // coalesced in t
  }
  __syncthreads();
  int cl = tid & 31, g2 = tid >> 5;
  #pragma unroll
  for (int i = 0; i < 4; ++i) {
    int tl2 = g2 + i * 8;
    embBf[(size_t)(b * Tc + t0 + tl2) * Cc + (c0 + cl)] =
        (__bf16)tile[cl][tl2];                                 // coalesced in c
  }
}

// ---------------------------------------------------------------------------
// Prep: row squared-norms of a bf16 matrix (rows x Cc); one wave per row
// ---------------------------------------------------------------------------
__global__ void rownorm_bf16(const __bf16* __restrict__ M,
                             float* __restrict__ out, int rows) {
  int wave = threadIdx.x >> 5, lane = threadIdx.x & 31;
  int row = blockIdx.x * (blockDim.x >> 5) + wave;
  if (row >= rows) return;
  const __bf16* p = M + (size_t)row * Cc;
  float s = 0.f;
  for (int i = lane; i < Cc; i += 32) { float v = (float)p[i]; s += v * v; }
  #pragma unroll
  for (int off = 16; off; off >>= 1) s += __shfl_xor(s, off, 32);
  if (lane == 0) out[row] = s;
}

// ---------------------------------------------------------------------------
// Main fused kernel: one workgroup (8 waves) per 16-token tile.
//   - dual bf16 WMMA GEMM vs W and codebook over all V=4096 columns
//   - online softmax for linear CE (logits never stored)
//   - d^2 tile (16 x 4096 f32) kept in LDS for margin + exact top-K (radix)
// ---------------------------------------------------------------------------
__global__ void __launch_bounds__(256)
main_loss_kernel(const __bf16* __restrict__ embBf,
                 const __bf16* __restrict__ Wbf,
                 const __bf16* __restrict__ CBbf,
                 const float*  __restrict__ x2g,
                 const float*  __restrict__ cb2,
                 const float*  __restrict__ bias,
                 const int*    __restrict__ codes,
                 float* __restrict__ outLin,
                 float* __restrict__ outMar,
                 float* __restrict__ outHn) {
  extern __shared__ char smem[];
  float*    d2s  = (float*)smem;                                   // 16*4096 f32
  __bf16*   As   = (__bf16*)(smem + 16 * Vc * 4);                  // 16*512 bf16
  unsigned* hist = (unsigned*)(smem + 16 * Vc * 4 + 16 * Cc * 2);  // 8*256 u32
  float*    redM = (float*)((char*)hist + 8 * 256 * 4);            // 8*16
  float*    redS = redM + 8 * 16;                                  // 8*16
  float*    rowM = redS + 8 * 16;                                  // 16
  float*    rowS = rowM + 16;                                      // 16
  float*    x2s  = rowS + 16;                                      // 16
  int*      labs = (int*)(x2s + 16);                               // 16

  const int tid  = threadIdx.x;
  const int wave = tid >> 5;
  const int lane = tid & 31;
  const int n0   = blockIdx.x * 16;

  // Stage A tile (16 x 512 bf16 = 16 KB), row norms, labels
  {
    const uint4* src = (const uint4*)(embBf + (size_t)n0 * Cc);
    uint4* dst = (uint4*)As;
    for (int i = tid; i < (16 * Cc * 2) / 16; i += 256) dst[i] = src[i];
    if (tid < 16) { x2s[tid] = x2g[n0 + tid]; labs[tid] = codes[n0 + tid]; }
  }
  __syncthreads();

  const int nIdx   = lane & 15;   // column-in-block / row-in-A per WMMA layout
  const int kh     = lane >> 4;   // k-half selector
  const int rowOff = kh * 8;      // C/D rows: VGPR r -> row r + 8*kh

  // Hoist all 16 A fragments (K chunks of 32) into registers: 128 VGPRs
  v16bf aReg[16];
  #pragma unroll
  for (int kc = 0; kc < 16; ++kc) {
    const __bf16* ar = As + nIdx * Cc + kc * 32 + kh * 16;
    v8bf lo = *(const v8bf*)(ar);
    v8bf hi = *(const v8bf*)(ar + 8);
    #pragma unroll
    for (int i = 0; i < 8; ++i) { aReg[kc][i] = lo[i]; aReg[kc][8 + i] = hi[i]; }
  }

  // Per-lane online softmax state: slot r covers row r + 8*kh, col = nIdx in block
  float m[8], s[8];
  #pragma unroll
  for (int r = 0; r < 8; ++r) { m[r] = -INFINITY; s[r] = 0.f; }

  // Sweep 16-wide column blocks of V; 32 blocks per wave
  for (int cb = wave; cb < Vc / 16; cb += 8) {
    const int col0 = cb * 16;
    const __bf16* wb  = Wbf  + (size_t)(col0 + nIdx) * Cc + kh * 16;
    const __bf16* cbp = CBbf + (size_t)(col0 + nIdx) * Cc + kh * 16;
    v8f accL = {};
    v8f accD = {};
    #pragma unroll
    for (int kc = 0; kc < 16; ++kc) {
      v16bf bw = *(const v16bf*)(wb  + kc * 32);
      v16bf bc = *(const v16bf*)(cbp + kc * 32);
      accL = __builtin_amdgcn_wmma_f32_16x16x32_bf16(
          false, aReg[kc], false, bw, (short)0, accL, false, false);
      accD = __builtin_amdgcn_wmma_f32_16x16x32_bf16(
          false, aReg[kc], false, bc, (short)0, accD, false, false);
    }
    const int col = col0 + nIdx;
    const float bb = bias[col];
    const float c2 = cb2[col];
    #pragma unroll
    for (int r = 0; r < 8; ++r) {
      const int row = r + rowOff;
      // linear logit -> online softmax
      float lg = accL[r] + bb;
      if (lg > m[r]) { s[r] = s[r] * __expf(m[r] - lg) + 1.f; m[r] = lg; }
      else           { s[r] += __expf(lg - m[r]); }
      // squared distance -> LDS
      float d2 = x2s[row] + c2 - 2.f * accD[r];
      d2s[row * Vc + col] = fmaxf(d2, 1e-12f);
    }
  }

  // Intra-wave softmax reduce over the 16 lanes sharing a row, then publish
  #pragma unroll
  for (int r = 0; r < 8; ++r) {
    #pragma unroll
    for (int off = 1; off < 16; off <<= 1) {
      float mo = __shfl_xor(m[r], off, 32);
      float so = __shfl_xor(s[r], off, 32);
      float mm = fmaxf(m[r], mo);
      s[r] = s[r] * __expf(m[r] - mm) + so * __expf(mo - mm);
      m[r] = mm;
    }
    if (nIdx == 0) {
      redM[wave * 16 + r + rowOff] = m[r];
      redS[wave * 16 + r + rowOff] = s[r];
    }
  }
  __syncthreads();
  if (wave == 0 && lane < 16) {
    float M = -INFINITY, S = 0.f;
    #pragma unroll
    for (int w = 0; w < 8; ++w) {
      float mo = redM[w * 16 + lane], so = redS[w * 16 + lane];
      float mm = fmaxf(M, mo);
      S = S * __expf(M - mm) + so * __expf(mo - mm);
      M = mm;
    }
    rowM[lane] = M; rowS[lane] = S;
  }
  __syncthreads();

  // Per-row phase: wave handles rows 2*wave and 2*wave+1
  for (int rr = 0; rr < 2; ++rr) {
    const int row = wave * 2 + rr;
    const int label = labs[row];
    const float* drow = d2s + row * Vc;
    const float d2lab = drow[label];
    const float distlab = sqrtf(d2lab);

    // --- margin: min distance over wrong codes ---
    float mn = INFINITY;
    for (int c = lane; c < Vc; c += 32) {
      float v = drow[c];
      if (c != label) mn = fminf(mn, v);
    }
    #pragma unroll
    for (int off = 16; off; off >>= 1) mn = fminf(mn, __shfl_xor(mn, off, 32));
    const float marg = fmaxf(0.f, distlab - sqrtf(mn) + 0.5f);

    // --- linear CE: recompute label logit (bf16 dot, f32 accumulate) ---
    {
      const __bf16* wrow = Wbf + (size_t)label * Cc;
      const __bf16* arow = As + row * Cc;
      float dot = 0.f;
      for (int i = lane; i < Cc; i += 32)
        dot += (float)arow[i] * (float)wrow[i];
      #pragma unroll
      for (int off = 16; off; off >>= 1) dot += __shfl_xor(dot, off, 32);
      if (lane == 0) {
        float lab_logit = dot + bias[label];
        outLin[n0 + row] = rowM[row] + __logf(rowS[row]) - lab_logit;
        outMar[n0 + row] = marg;
      }
    }

    // --- exact Kth-smallest d^2 via 4-pass byte radix select on float bits ---
    unsigned prefix = 0;
    int remaining = Kc;
    unsigned* h = hist + wave * 256;
    for (int shift = 24; shift >= 0; shift -= 8) {
      for (int i = lane; i < 256; i += 32) h[i] = 0u;
      __builtin_amdgcn_wave_barrier();
      for (int c = lane; c < Vc; c += 32) {
        unsigned u = __float_as_uint(drow[c]);   // positive: order-preserving
        bool match = (shift == 24) || ((u >> (shift + 8)) == (prefix >> (shift + 8)));
        if (match) atomicAdd(&h[(u >> shift) & 255u], 1u);
      }
      __builtin_amdgcn_wave_barrier();
      int chosen = 0, rem2 = remaining;
      if (lane == 0) {
        int acc = 0;
        for (int bin = 0; bin < 256; ++bin) {
          int hv = (int)h[bin];
          if (acc + hv >= remaining) { chosen = bin; rem2 = remaining - acc; break; }
          acc += hv;
        }
      }
      chosen = __shfl(chosen, 0, 32);
      rem2   = __shfl(rem2, 0, 32);
      prefix |= ((unsigned)chosen) << shift;
      remaining = rem2;
    }
    const float theta = __uint_as_float(prefix);  // 100th smallest d^2

    // --- hard-negative CE over top-K set (with label swap-in) ---
    float se = 0.f;
    for (int c = lane; c < Vc; c += 32) {
      float v = drow[c];
      if (v <= theta) se += __expf(-sqrtf(v));
    }
    #pragma unroll
    for (int off = 16; off; off >>= 1) se += __shfl_xor(se, off, 32);
    if (lane == 0) {
      bool in_topk = (d2lab <= theta);
      float S = in_topk ? se : (se - __expf(-sqrtf(theta)) + __expf(-distlab));
      outHn[n0 + row] = distlab + __logf(fmaxf(S, 1e-30f));
    }
  }
}

// ---------------------------------------------------------------------------
// Final: mean of each per-row loss, weighted total
// ---------------------------------------------------------------------------
__global__ void final_reduce(const float* __restrict__ lin,
                             const float* __restrict__ mar,
                             const float* __restrict__ hn,
                             float* __restrict__ out) {
  __shared__ float sl[256], sm[256], sh[256];
  int tid = threadIdx.x;
  float a = 0.f, b = 0.f, c = 0.f;
  for (int i = tid; i < Nc; i += 256) { a += lin[i]; b += mar[i]; c += hn[i]; }
  sl[tid] = a; sm[tid] = b; sh[tid] = c;
  __syncthreads();
  for (int off = 128; off; off >>= 1) {
    if (tid < off) {
      sl[tid] += sl[tid + off];
      sm[tid] += sm[tid + off];
      sh[tid] += sh[tid + off];
    }
    __syncthreads();
  }
  if (tid == 0)
    out[0] = (1.0f * sl[0] + 0.5f * sm[0] + 0.5f * sh[0]) / (float)Nc;
}

// ---------------------------------------------------------------------------
extern "C" void kernel_launch(void* const* d_in, const int* in_sizes, int n_in,
                              void* d_out, int out_size, void* d_ws, size_t ws_size,
                              hipStream_t stream) {
  (void)in_sizes; (void)n_in; (void)out_size; (void)ws_size;
  const float* student  = (const float*)d_in[0];  // (B, C, T) f32
  const int*   codes    = (const int*)  d_in[1];  // (B, T)
  const float* codebook = (const float*)d_in[2];  // (V, C) f32
  const float* W        = (const float*)d_in[3];  // (V, C) f32
  const float* bias     = (const float*)d_in[4];  // (V,) f32

  char* ws = (char*)d_ws;
  size_t off = 0;
  __bf16* Wbf   = (__bf16*)(ws + off); off += (size_t)Vc * Cc * 2;
  __bf16* CBbf  = (__bf16*)(ws + off); off += (size_t)Vc * Cc * 2;
  __bf16* embBf = (__bf16*)(ws + off); off += (size_t)Nc * Cc * 2;
  float* x2  = (float*)(ws + off); off += (size_t)Nc * 4;
  float* cb2 = (float*)(ws + off); off += (size_t)Vc * 4;
  float* oL  = (float*)(ws + off); off += (size_t)Nc * 4;
  float* oM  = (float*)(ws + off); off += (size_t)Nc * 4;
  float* oH  = (float*)(ws + off); off += (size_t)Nc * 4;

  prep_convert<<<(Vc * Cc + 255) / 256, 256, 0, stream>>>(W, codebook, Wbf, CBbf);
  dim3 tg(Tc / 32, Cc / 32, Bc);
  prep_transpose<<<tg, 256, 0, stream>>>(student, embBf);
  rownorm_bf16<<<Vc / 8, 256, 0, stream>>>(CBbf, cb2, Vc);
  rownorm_bf16<<<Nc / 8, 256, 0, stream>>>(embBf, x2, Nc);

  const size_t lds = (size_t)16 * Vc * 4      // d2 tile
                   + (size_t)16 * Cc * 2      // A tile
                   + (size_t)8 * 256 * 4      // radix histograms
                   + (size_t)(8 * 16 * 2) * 4 // redM/redS
                   + (size_t)(16 * 2) * 4     // rowM/rowS
                   + (size_t)16 * 4           // x2s
                   + (size_t)16 * 4;          // labels
  main_loss_kernel<<<Nc / 16, 256, lds, stream>>>(embBf, Wbf, CBbf, x2, cb2,
                                                  bias, codes, oL, oM, oH);
  final_reduce<<<1, 256, 0, stream>>>(oL, oM, oH, (float*)d_out);
}
